// RSSMCore_37718402793717
// MI455X (gfx1250) — compile-verified, matrix operand order checked by reading, not run on hardware
//
#include <hip/hip_runtime.h>
#include <math.h>

// ---------------- types ----------------
typedef __bf16 bf16;
typedef __attribute__((ext_vector_type(16))) __bf16 v16bf;
typedef __attribute__((ext_vector_type(8)))  __bf16 v8bf;
typedef __attribute__((ext_vector_type(8)))  float  v8f;

// Problem constants: N=64 steps, B=256, E=1536, A=32, D=2048, S=64, H=2048
#define NSTEP 64
#define BATCH 256
#define EDIM  1536
#define ADIM  32
#define DDIM  2048
#define SDIM  64
#define HDIM  2048
#define G3    6144   // 3*D
#define KZA   96     // S + A
#define KHE   3584   // D + E

__device__ __forceinline__ bf16 f2bf(float f) {
  unsigned u = __builtin_bit_cast(unsigned, f);
  unsigned r = (u + 0x7FFFu + ((u >> 16) & 1u)) >> 16;
  unsigned short h = (unsigned short)r;
  return __builtin_bit_cast(bf16, h);
}

__device__ __forceinline__ float eluf(float x) {
  return x > 0.f ? x : (expf(x) - 1.f);
}

// ---------------- WMMA GEMM ----------------
// C[M,N] = act( A[M,K](bf16) * Wt[N,K]^T(bf16) + bias[N] )
// MODE 0: store f32, no activation.  MODE 1: store bf16 with ELU.
// Block: 256 threads = 8 waves arranged 2(M) x 4(N).
// Block tile 64(M) x 256(N); wave tile 32(M) x 64(N).
// K is consumed in 64-wide chunks: A (64x64 = 8KB) and B (256x64 = 32KB)
// tiles live in double-buffered LDS (80KB total), filled with
// global_load_async_to_lds_b128 (ASYNCcnt) while 16 WMMAs/wave consume the
// previous chunk -> one barrier per 64 K-elements, no fragment registers
// live across iterations, no spills.
template<int MODE>
__global__ void __launch_bounds__(256, 1)
gemm_wmma(const bf16* __restrict__ A, const bf16* __restrict__ Wt,
          const float* __restrict__ bias, void* __restrict__ outv,
          int M, int N, int K, int ldc) {
  __shared__ __align__(16) bf16 smA[2][64 * 64];
  __shared__ __align__(16) bf16 smB[2][256 * 64];

  const int lane  = threadIdx.x & 31;
  const int wave  = threadIdx.x >> 5;
  const int waveM = wave >> 2;            // 0..1
  const int waveN = wave & 3;             // 0..3
  const int mBlk  = blockIdx.y * 64;
  const int nBlk  = blockIdx.x * 256;
  const int mBase = mBlk + waveM * 32;
  const int nBase = nBlk + waveN * 64;
  // wave-uniform predicate in an SGPR -> scalar branch around WMMA
  const int active = __builtin_amdgcn_readfirstlane((nBase < N) ? 1 : 0);

  auto asyncB128 = [&](const bf16* ldsp, const bf16* gp) {
    unsigned lds = (unsigned)(uintptr_t)ldsp;
    unsigned long long ga = (unsigned long long)(uintptr_t)gp;
    asm volatile("global_load_async_to_lds_b128 %0, %1, off"
                 :: "v"(lds), "v"(ga) : "memory");
  };

  // Stage A (64x64) + B (256x64) tiles for chunk k0 into LDS buffer buf.
  // Per wave: 2 + 8 async b128 instructions (each moves 32 lanes x 16B).
  auto stage64 = [&](int k0, int buf) {
    const int row = threadIdx.x >> 2;     // 0..63
    const int q   = threadIdx.x & 3;      // base 16B chunk in a 128B row
#pragma unroll
    for (int j = 0; j < 2; ++j) {
      const int qw = q + j * 4;           // 0..7
      asyncB128(&smA[buf][row * 64 + qw * 8],
                A + (size_t)(mBlk + row) * K + k0 + qw * 8);
    }
#pragma unroll
    for (int j = 0; j < 8; ++j) {
      const int br = row + (j >> 1) * 64; // 0..255
      const int qw = q + (j & 1) * 4;     // 0..7
      int gr = nBlk + br;
      if (gr > N - 1) gr = N - 1;         // clamp: stay inside Wt for small N
      asyncB128(&smB[buf][br * 64 + qw * 8],
                Wt + (size_t)gr * K + k0 + qw * 8);
    }
  };

  auto barrier = [&]() {
    asm volatile("s_wait_asynccnt 0x0\n\t"
                 "s_barrier_signal -1\n\t"
                 "s_barrier_wait -1" ::: "memory");
  };

  // Read one 16x32 fragment from an LDS tile (rows of 64 bf16), K-substep sub.
  // Lanes 0-15: K {0..7,16..23}; lanes 16-31: K {8..15,24..31}.
  auto frag = [&](const bf16* tile, int rowBase, int sub) -> v16bf {
    const bf16* p = tile + (rowBase + (lane & 15)) * 64 + sub * 32 + ((lane & 16) >> 1);
    v8bf c0 = *(const v8bf*)p;
    v8bf c1 = *(const v8bf*)(p + 16);
    return __builtin_shufflevector(c0, c1, 0,1,2,3,4,5,6,7,8,9,10,11,12,13,14,15);
  };

  v8f acc[2][4];
#pragma unroll
  for (int mi = 0; mi < 2; ++mi)
#pragma unroll
    for (int ni = 0; ni < 4; ++ni) acc[mi][ni] = (v8f){};

  // ---- prologue ----
  stage64(0, 0);
  barrier();

  // ---- pipelined K loop (64-wide chunks) ----
  const int nChunk = (K + 63) >> 6;
  int cur = 0;
  for (int c = 0;;) {
    const bool hasNext = (c + 1) < nChunk;
    const int nxt = cur ^ 1;
    if (hasNext) stage64((c + 1) * 64, nxt);  // async: overlaps WMMAs below

    auto compute = [&](int sub) {
      v16bf af0 = frag(smA[cur], waveM * 32,      sub);
      v16bf af1 = frag(smA[cur], waveM * 32 + 16, sub);
      v16bf b0  = frag(smB[cur], waveN * 64,      sub);
      v16bf b1  = frag(smB[cur], waveN * 64 + 16, sub);
      v16bf b2  = frag(smB[cur], waveN * 64 + 32, sub);
      v16bf b3  = frag(smB[cur], waveN * 64 + 48, sub);
      acc[0][0] = __builtin_amdgcn_wmma_f32_16x16x32_bf16(false, af0, false, b0, (short)0, acc[0][0], false, false);
      acc[0][1] = __builtin_amdgcn_wmma_f32_16x16x32_bf16(false, af0, false, b1, (short)0, acc[0][1], false, false);
      acc[0][2] = __builtin_amdgcn_wmma_f32_16x16x32_bf16(false, af0, false, b2, (short)0, acc[0][2], false, false);
      acc[0][3] = __builtin_amdgcn_wmma_f32_16x16x32_bf16(false, af0, false, b3, (short)0, acc[0][3], false, false);
      acc[1][0] = __builtin_amdgcn_wmma_f32_16x16x32_bf16(false, af1, false, b0, (short)0, acc[1][0], false, false);
      acc[1][1] = __builtin_amdgcn_wmma_f32_16x16x32_bf16(false, af1, false, b1, (short)0, acc[1][1], false, false);
      acc[1][2] = __builtin_amdgcn_wmma_f32_16x16x32_bf16(false, af1, false, b2, (short)0, acc[1][2], false, false);
      acc[1][3] = __builtin_amdgcn_wmma_f32_16x16x32_bf16(false, af1, false, b3, (short)0, acc[1][3], false, false);
    };

    if (active) {
      compute(0);
      if (c * 64 + 32 < K) compute(1);    // uniform: skipped only on a 32-wide tail
    }
    if (!hasNext) break;
    barrier();                            // own async done + block rendezvous
    cur = nxt;
    ++c;
  }

  if (!active) return;

  // Epilogue. C/D layout: VGPR r -> M = r (lanes 0-15) / r+8 (lanes 16-31), N = lane&15.
  const int col  = nBase + (lane & 15);
  const int rOff = (lane >> 4) << 3;
  const float bv0 = bias[col];
  const float bv1 = bias[col + 16];
  const float bv2 = bias[col + 32];
  const float bv3 = bias[col + 48];

  float* outF = (float*)outv;
  bf16*  outB = (bf16*)outv;

#pragma unroll
  for (int mi = 0; mi < 2; ++mi) {
#pragma unroll
    for (int r = 0; r < 8; ++r) {
      const size_t row = (size_t)(mBase + mi * 16 + rOff + r);
      float v0 = acc[mi][0][r] + bv0;
      float v1 = acc[mi][1][r] + bv1;
      float v2 = acc[mi][2][r] + bv2;
      float v3 = acc[mi][3][r] + bv3;
      if constexpr (MODE == 0) {
        outF[row * ldc + col     ] = v0;
        outF[row * ldc + col + 16] = v1;
        outF[row * ldc + col + 32] = v2;
        outF[row * ldc + col + 48] = v3;
      } else {
        outB[row * ldc + col     ] = f2bf(eluf(v0));
        outB[row * ldc + col + 16] = f2bf(eluf(v1));
        outB[row * ldc + col + 32] = f2bf(eluf(v2));
        outB[row * ldc + col + 48] = f2bf(eluf(v3));
      }
    }
  }
}

// ---------------- prep / elementwise kernels ----------------

// dst[n*dstStride + colOff + k] = bf16(src[k*N + n])   (transpose + convert)
__global__ void tconv(const float* __restrict__ src, bf16* __restrict__ dst,
                      int K, int N, int dstStride, int colOff) {
  int i = blockIdx.x * blockDim.x + threadIdx.x;
  if (i >= K * N) return;
  int k = i / N, n = i - k * N;
  dst[(size_t)n * dstStride + colOff + k] = f2bf(src[i]);
}

__global__ void copy_f32(const float* __restrict__ src, float* __restrict__ dst, int n) {
  int i = blockIdx.x * blockDim.x + threadIdx.x;
  if (i < n) dst[i] = src[i];
}

// h *= (1 - reset); also emit bf16 copy for the gh GEMM
__global__ void mask_h(float* __restrict__ hA, bf16* __restrict__ hbf,
                       const unsigned char* __restrict__ reset) {
  int i = blockIdx.x * blockDim.x + threadIdx.x;   // 256*2048 threads
  int b = i >> 11;
  float rm = reset[b] ? 0.f : 1.f;
  float h = hA[i] * rm;
  hA[i] = h;
  hbf[i] = f2bf(h);
}

// zaA[b, 0:64]  = bf16(z[b]*rm);  zaA[b, 64:96] = bf16(action_t[b])
__global__ void build_za(const float* __restrict__ zA, const float* __restrict__ action,
                         const unsigned char* __restrict__ reset, bf16* __restrict__ zaA) {
  int i = blockIdx.x * blockDim.x + threadIdx.x;   // 256*96 threads
  if (i >= BATCH * KZA) return;
  int b = i / KZA, k = i - b * KZA;
  float rm = reset[b] ? 0.f : 1.f;
  float v = (k < SDIM) ? zA[b * SDIM + k] * rm : action[b * ADIM + (k - SDIM)];
  zaA[i] = f2bf(v);
}

// GRU gate math + h_new; writes hA (f32 carry), hbf (bf16), hE[:,0:2048], features[t]
__global__ void gru_update(const float* __restrict__ gx, const float* __restrict__ gh,
                           float* __restrict__ hA, bf16* __restrict__ hbf,
                           bf16* __restrict__ hE, float* __restrict__ features, int t) {
  int i = blockIdx.x * blockDim.x + threadIdx.x;   // 256*2048 threads
  int b = i >> 11, d = i & 2047;
  size_t g = (size_t)b * G3 + d;
  float xr = gx[g], xu = gx[g + DDIM], xn = gx[g + 2 * DDIM];
  float hr = gh[g], hu = gh[g + DDIM], hn = gh[g + 2 * DDIM];
  float h = hA[i];
  float r = 1.f / (1.f + expf(-(xr + hr)));
  float u = 1.f / (1.f + expf(-(xu + hu)));
  float c = tanhf(xn + r * hn);
  float hnew = (1.f - u) * c + u * h;
  hA[i] = hnew;
  hbf[i] = f2bf(hnew);
  hE[(size_t)b * KHE + d] = f2bf(hnew);
  features[((size_t)(t * BATCH + b)) * (DDIM + SDIM) + d] = hnew;
}

// hE[:, 2048:3584] = bf16(embed_t)
__global__ void embed_conv(const float* __restrict__ e, bf16* __restrict__ hE) {
  int i = blockIdx.x * blockDim.x + threadIdx.x;   // 256*1536 threads
  int b = i / EDIM, j = i - b * EDIM;
  hE[(size_t)b * KHE + DDIM + j] = f2bf(e[i]);
}

// sample = mean + (softplus(std_raw)+0.1)*noise; write samples, features tail, z carry
__global__ void rsample_k(const float* __restrict__ post, const float* __restrict__ noise,
                          float* __restrict__ samples, float* __restrict__ features,
                          float* __restrict__ zA, int t) {
  int i = blockIdx.x * blockDim.x + threadIdx.x;   // 256*64 threads
  int b = i >> 6, s = i & 63;
  float mean = post[b * 2 * SDIM + s];
  float sr   = post[b * 2 * SDIM + SDIM + s];
  float sp   = (sr > 20.f) ? sr : log1pf(expf(sr));
  float smp  = mean + (sp + 0.1f) * noise[i];
  samples[i] = smp;
  zA[i] = smp;
  features[((size_t)(t * BATCH + b)) * (DDIM + SDIM) + DDIM + s] = smp;
}

// ---------------- host driver ----------------
extern "C" void kernel_launch(void* const* d_in, const int* in_sizes, int n_in,
                              void* d_out, int out_size, void* d_ws, size_t ws_size,
                              hipStream_t stream) {
  (void)in_sizes; (void)n_in; (void)out_size; (void)ws_size;

  const float* embed      = (const float*)d_in[0];
  const float* action     = (const float*)d_in[1];
  const unsigned char* reset = (const unsigned char*)d_in[2];
  const float* h0         = (const float*)d_in[3];
  const float* z0         = (const float*)d_in[4];
  const float* noise      = (const float*)d_in[5];
  const float* z_mlp_w    = (const float*)d_in[6];
  const float* z_mlp_b    = (const float*)d_in[7];
  const float* a_mlp_w    = (const float*)d_in[8];
  const float* gru_wih    = (const float*)d_in[9];
  const float* gru_whh    = (const float*)d_in[10];
  const float* gru_bih    = (const float*)d_in[11];
  const float* gru_bhh    = (const float*)d_in[12];
  const float* post_h_w   = (const float*)d_in[13];
  const float* post_h_b   = (const float*)d_in[14];
  const float* post_e_w   = (const float*)d_in[15];
  const float* post_w     = (const float*)d_in[16];
  const float* post_b     = (const float*)d_in[17];
  const float* prior_h_w  = (const float*)d_in[18];
  const float* prior_h_b  = (const float*)d_in[19];
  const float* prior_w    = (const float*)d_in[20];
  const float* prior_b    = (const float*)d_in[21];

  // d_out slices (return order): priors, posts, samples, features, hT, zT
  float* out      = (float*)d_out;
  float* priors   = out;
  float* posts    = priors  + (size_t)NSTEP * BATCH * 2 * SDIM;
  float* samples  = posts   + (size_t)NSTEP * BATCH * 2 * SDIM;
  float* features = samples + (size_t)NSTEP * BATCH * SDIM;
  float* hT       = features + (size_t)NSTEP * BATCH * (DDIM + SDIM);
  float* zT       = hT + (size_t)BATCH * DDIM;

  // workspace carve-out
  char* p = (char*)d_ws;
  auto alloc = [&](size_t bytes) -> char* {
    char* r = p; p += (bytes + 255) & ~(size_t)255; return r;
  };
  bf16* WzaT    = (bf16*)alloc((size_t)HDIM * KZA  * 2);  // [2048, 96]
  bf16* WihT    = (bf16*)alloc((size_t)G3   * HDIM * 2);  // [6144, 2048]
  bf16* WhhT    = (bf16*)alloc((size_t)G3   * DDIM * 2);  // [6144, 2048]
  bf16* WpT     = (bf16*)alloc((size_t)HDIM * KHE  * 2);  // [2048, 3584]
  bf16* PostWT  = (bf16*)alloc((size_t)2*SDIM * HDIM * 2);// [128, 2048]
  bf16* PriorHT = (bf16*)alloc((size_t)HDIM * DDIM * 2);  // [2048, 2048]
  bf16* PriorWT = (bf16*)alloc((size_t)2*SDIM * HDIM * 2);// [128, 2048]
  float* hA     = (float*)alloc((size_t)BATCH * DDIM * 4);
  float* zA     = (float*)alloc((size_t)BATCH * SDIM * 4);
  bf16* hbf     = (bf16*)alloc((size_t)BATCH * DDIM * 2);
  bf16* zaA     = (bf16*)alloc((size_t)BATCH * KZA  * 2);
  bf16* zaOut   = (bf16*)alloc((size_t)BATCH * HDIM * 2);
  float* gx     = (float*)alloc((size_t)BATCH * G3 * 4);
  float* gh     = (float*)alloc((size_t)BATCH * G3 * 4);
  bf16* hE      = (bf16*)alloc((size_t)BATCH * KHE * 2);
  bf16* postIn  = (bf16*)alloc((size_t)BATCH * HDIM * 2);
  bf16* priorMid= (bf16*)alloc((size_t)BATCH * HDIM * 2);

  auto blocks = [](int n) { return (n + 255) / 256; };

  // ---- one-time weight transpose + bf16 convert ----
  tconv<<<blocks(SDIM * HDIM), 256, 0, stream>>>(z_mlp_w,  WzaT, SDIM, HDIM, KZA, 0);
  tconv<<<blocks(ADIM * HDIM), 256, 0, stream>>>(a_mlp_w,  WzaT, ADIM, HDIM, KZA, SDIM);
  tconv<<<blocks(HDIM * G3),   256, 0, stream>>>(gru_wih,  WihT, HDIM, G3, HDIM, 0);
  tconv<<<blocks(DDIM * G3),   256, 0, stream>>>(gru_whh,  WhhT, DDIM, G3, DDIM, 0);
  tconv<<<blocks(DDIM * HDIM), 256, 0, stream>>>(post_h_w, WpT,  DDIM, HDIM, KHE, 0);
  tconv<<<blocks(EDIM * HDIM), 256, 0, stream>>>(post_e_w, WpT,  EDIM, HDIM, KHE, DDIM);
  tconv<<<blocks(HDIM * 2*SDIM),256,0, stream>>>(post_w,   PostWT, HDIM, 2*SDIM, HDIM, 0);
  tconv<<<blocks(DDIM * HDIM), 256, 0, stream>>>(prior_h_w,PriorHT, DDIM, HDIM, DDIM, 0);
  tconv<<<blocks(HDIM * 2*SDIM),256,0, stream>>>(prior_w,  PriorWT, HDIM, 2*SDIM, HDIM, 0);

  // ---- init carry ----
  copy_f32<<<blocks(BATCH * DDIM), 256, 0, stream>>>(h0, hA, BATCH * DDIM);
  copy_f32<<<blocks(BATCH * SDIM), 256, 0, stream>>>(z0, zA, BATCH * SDIM);

  auto gemm = [&](const bf16* A, const bf16* Wt, const float* bias, void* o,
                  int N, int K, int ldc, int mode) {
    dim3 grid((N + 255) / 256, BATCH / 64);
    if (mode == 0)
      gemm_wmma<0><<<grid, 256, 0, stream>>>(A, Wt, bias, o, BATCH, N, K, ldc);
    else
      gemm_wmma<1><<<grid, 256, 0, stream>>>(A, Wt, bias, o, BATCH, N, K, ldc);
  };

  // ---- sequential scan over timesteps ----
  for (int t = 0; t < NSTEP; ++t) {
    const unsigned char* rst = reset + (size_t)t * BATCH;
    mask_h  <<<blocks(BATCH * DDIM), 256, 0, stream>>>(hA, hbf, rst);
    build_za<<<blocks(BATCH * KZA),  256, 0, stream>>>(zA, action + (size_t)t * BATCH * ADIM, rst, zaA);

    // za = elu(z@Wz + a@Wa + b)                 [256,96]x[96,2048] -> bf16
    gemm(zaA, WzaT, z_mlp_b, zaOut, HDIM, KZA, HDIM, 1);
    // gx = za@Wih + bih,  gh = h@Whh + bhh      [256,2048]x[2048,6144] -> f32
    gemm(zaOut, WihT, gru_bih, gx, G3, HDIM, G3, 0);
    gemm(hbf,   WhhT, gru_bhh, gh, G3, DDIM, G3, 0);

    gru_update<<<blocks(BATCH * DDIM), 256, 0, stream>>>(gx, gh, hA, hbf, hE, features, t);
    embed_conv<<<blocks(BATCH * EDIM), 256, 0, stream>>>(embed + (size_t)t * BATCH * EDIM, hE);

    // post_in = elu([h_new|e]@[Wph;Wpe] + b)    [256,3584]x[3584,2048] -> bf16
    gemm(hE, WpT, post_h_b, postIn, HDIM, KHE, HDIM, 1);
    // post = post_in@Wp + b  -> directly into d_out posts[t]
    gemm(postIn, PostWT, post_b, posts + (size_t)t * BATCH * 2 * SDIM, 2 * SDIM, HDIM, 2 * SDIM, 0);

    rsample_k<<<blocks(BATCH * SDIM), 256, 0, stream>>>(
        posts + (size_t)t * BATCH * 2 * SDIM, noise + (size_t)t * BATCH * SDIM,
        samples + (size_t)t * BATCH * SDIM, features, zA, t);

    // prior = elu(h_new@Wprh + b)@Wpr + b  -> directly into d_out priors[t]
    gemm(hbf, PriorHT, prior_h_b, priorMid, HDIM, DDIM, HDIM, 1);
    gemm(priorMid, PriorWT, prior_b, priors + (size_t)t * BATCH * 2 * SDIM, 2 * SDIM, HDIM, 2 * SDIM, 0);
  }

  // ---- final carry outputs ----
  copy_f32<<<blocks(BATCH * DDIM), 256, 0, stream>>>(hA, hT, BATCH * DDIM);
  copy_f32<<<blocks(BATCH * SDIM), 256, 0, stream>>>(zA, zT, BATCH * SDIM);
}
